// GAT_85779086835952
// MI455X (gfx1250) — matrix operand order repaired and checked by reading, not compiled, not run on hardware
//
#include <hip/hip_runtime.h>
#include <hip/hip_bf16.h>
#include <stdint.h>

// ---------- WMMA types ----------
typedef __attribute__((ext_vector_type(16))) __bf16 v16bf;
typedef __attribute__((ext_vector_type(8)))  float  v8f;

union FragAB {
  uint4  u4[2];
  v16bf  v;
};

__device__ __forceinline__ unsigned short f32_to_bf16_rne(float f) {
  unsigned u = __float_as_uint(f);
  u += 0x7FFFu + ((u >> 16) & 1u);
  return (unsigned short)(u >> 16);
}

// order-preserving float<->uint encoding for atomicMax-based segment max
__device__ __forceinline__ unsigned ordered_enc(float f) {
  unsigned u = __float_as_uint(f);
  return (u & 0x80000000u) ? ~u : (u | 0x80000000u);
}
__device__ __forceinline__ float ordered_dec(unsigned k) {
  unsigned b = (k & 0x80000000u) ? (k & 0x7FFFFFFFu) : ~k;
  return __uint_as_float(b);
}

#define CDIV(a, b) (((a) + (b) - 1) / (b))

// ---------- conversion kernels ----------
__global__ __launch_bounds__(256) void k_f32_to_bf16(const float* __restrict__ in,
                                                     unsigned short* __restrict__ out, int n) {
  int i = blockIdx.x * blockDim.x + threadIdx.x;
  if (i < n) out[i] = f32_to_bf16_rne(in[i]);
}

// W: [K, Nc] row-major f32  ->  WT: [Nc, K] row-major bf16
__global__ __launch_bounds__(256) void k_transpose_bf16(const float* __restrict__ W,
                                                        unsigned short* __restrict__ WT,
                                                        int K, int Nc) {
  int i = blockIdx.x * blockDim.x + threadIdx.x;
  if (i >= K * Nc) return;
  int k = i / Nc, j = i % Nc;
  WT[(size_t)j * K + k] = f32_to_bf16_rne(W[i]);
}

// ---------- WMMA GEMM: C[M,Nc] = A[M,K](bf16) * B[K,Nc], B given as BT[Nc,K](bf16) ----------
__global__ __launch_bounds__(128) void k_gemm_wmma_bf16(const unsigned short* __restrict__ A,
                                                        const unsigned short* __restrict__ BT,
                                                        float* __restrict__ C,
                                                        int M, int Nc, int K) {
  const int lane = threadIdx.x & 31;
  const int wave = threadIdx.x >> 5;
  const int tiles_n = Nc >> 4;
  const int total = (M >> 4) * tiles_n;
  const int tile = blockIdx.x * (blockDim.x >> 5) + wave;
  if (tile >= total) return;
  const int ti = tile / tiles_n;
  const int tj = tile % tiles_n;
  const int r15 = lane & 15;       // A row within tile / B column within tile
  const int hi  = lane >> 4;

  const unsigned short* arow = A  + (size_t)(ti * 16 + r15) * K;
  const unsigned short* brow = BT + (size_t)(tj * 16 + r15) * K;

  v8f acc = {};
  for (int kk = 0; kk < K; kk += 32) {
    FragAB a, b;
    // A 16x32 bf16: lane holds K[8*hi .. 8*hi+7] and K[16+8*hi .. 16+8*hi+7]
    a.u4[0] = *(const uint4*)(arow + kk + 8 * hi);
    a.u4[1] = *(const uint4*)(arow + kk + 16 + 8 * hi);
    // B 32x16 bf16: lane holds column, K[8*hi .. 8*hi+7] and K[16+8*hi .. 16+8*hi+7] in W^T
    b.u4[0] = *(const uint4*)(brow + kk + 8 * hi);
    b.u4[1] = *(const uint4*)(brow + kk + 16 + 8 * hi);
    acc = __builtin_amdgcn_wmma_f32_16x16x32_bf16(false, a.v, false, b.v,
                                                  (short)0, acc, false, false);
  }
  // C/D layout: lane n = r15, VGPR r -> row r + 8*hi
  float* crow = C + (size_t)(ti * 16) * Nc + tj * 16 + r15;
#pragma unroll
  for (int r = 0; r < 8; ++r)
    crow[(size_t)(r + 8 * hi) * Nc] = acc[r];
}

// ---------- attention score dots: as/ad[n,h] = sum_c xp[n,h,c]*att[h,c] ----------
__global__ __launch_bounds__(256) void k_att_scores(const float* __restrict__ xp,
                                                    const float* __restrict__ att_s,
                                                    const float* __restrict__ att_d,
                                                    float* __restrict__ as_out,
                                                    float* __restrict__ ad_out,
                                                    int Nn, int H, int Cc) {
  int i = blockIdx.x * blockDim.x + threadIdx.x;
  if (i >= Nn * H) return;
  int h = i % H;
  const float* row = xp + (size_t)i * Cc;          // (n*H + h) * Cc
  const float* ws = att_s + h * Cc;
  const float* wd = att_d + h * Cc;
  float ss = 0.f, sd = 0.f;
  for (int c = 0; c < Cc; ++c) { float v = row[c]; ss += v * ws[c]; sd += v * wd[c]; }
  as_out[i] = ss; ad_out[i] = sd;
}

// ---------- edge phase ----------
__global__ __launch_bounds__(256) void k_edge_logits_max(const int* __restrict__ ei, int E, int Nn, int H,
                                                         const float* __restrict__ as_n,
                                                         const float* __restrict__ ad_n,
                                                         float* __restrict__ alpha,
                                                         unsigned* __restrict__ mkey) {
  int e = blockIdx.x * blockDim.x + threadIdx.x;
  if (e >= E + Nn) return;
  int s, d;
  if (e < E) { s = ei[e]; d = ei[E + e]; } else { s = d = e - E; }
  for (int h = 0; h < H; ++h) {
    float v = as_n[s * H + h] + ad_n[d * H + h];
    v = v > 0.f ? v : 0.2f * v;                     // leaky_relu(0.2)
    alpha[(size_t)e * H + h] = v;
    atomicMax(&mkey[d * H + h], ordered_enc(v));
  }
}

__global__ __launch_bounds__(256) void k_decode_max(const unsigned* __restrict__ mkey,
                                                    float* __restrict__ mdec, int n) {
  int i = blockIdx.x * blockDim.x + threadIdx.x;
  if (i >= n) return;
  float f = ordered_dec(mkey[i]);
  unsigned b = __float_as_uint(f);
  if (((b >> 23) & 0xFFu) == 0xFFu) f = 0.f;        // !isfinite -> 0 (matches reference)
  mdec[i] = f;
}

__global__ __launch_bounds__(256) void k_edge_exp_sum(const int* __restrict__ ei, int E, int Nn, int H,
                                                      const float* __restrict__ mdec,
                                                      float* __restrict__ alpha,
                                                      float* __restrict__ ssum) {
  int e = blockIdx.x * blockDim.x + threadIdx.x;
  if (e >= E + Nn) return;
  int d = (e < E) ? ei[E + e] : (e - E);
  for (int h = 0; h < H; ++h) {
    float ex = __expf(alpha[(size_t)e * H + h] - mdec[d * H + h]);
    alpha[(size_t)e * H + h] = ex;
    atomicAdd(&ssum[d * H + h], ex);
  }
}

__global__ __launch_bounds__(256) void k_edge_norm(const int* __restrict__ ei, int E, int Nn, int H,
                                                   const float* __restrict__ ssum,
                                                   float* __restrict__ alpha) {
  long long i = (long long)blockIdx.x * blockDim.x + threadIdx.x;
  long long total = (long long)(E + Nn) * H;
  if (i >= total) return;
  int e = (int)(i / H), h = (int)(i % H);
  int d = (e < E) ? ei[E + e] : (e - E);
  alpha[i] = alpha[i] / (ssum[d * H + h] + 1e-16f);
}

// out[d, :] += alpha[e,h] * feat[s, :], vectorized float4 over H*C channels
__global__ __launch_bounds__(256) void k_edge_aggregate(const int* __restrict__ ei, int E, int Nn,
                                                        int H, int C,
                                                        const float* __restrict__ alpha,
                                                        const float* __restrict__ feat,
                                                        float* __restrict__ out) {
  const int hc4 = (H * C) >> 2;
  long long i = (long long)blockIdx.x * blockDim.x + threadIdx.x;
  long long total = (long long)(E + Nn) * hc4;
  if (i >= total) return;
  int e = (int)(i / hc4);
  int c4 = ((int)(i % hc4)) << 2;
  int h = c4 / C;
  int s, d;
  if (e < E) { s = ei[e]; d = ei[E + e]; } else { s = d = e - E; }
  float w = alpha[(size_t)e * H + h];
  const float4 f = *(const float4*)(feat + (size_t)s * H * C + c4);
  float* o = out + (size_t)d * H * C + c4;
  atomicAdd(o + 0, w * f.x);
  atomicAdd(o + 1, w * f.y);
  atomicAdd(o + 2, w * f.z);
  atomicAdd(o + 3, w * f.w);
}

// h = elu(agg + bias), store as bf16 for next GEMM
__global__ __launch_bounds__(256) void k_bias_elu_bf16(const float* __restrict__ agg,
                                                       const float* __restrict__ bias,
                                                       unsigned short* __restrict__ hout,
                                                       int Nn, int C) {
  int i = blockIdx.x * blockDim.x + threadIdx.x;
  if (i >= Nn * C) return;
  float v = agg[i] + bias[i % C];
  v = v > 0.f ? v : (__expf(v) - 1.f);
  hout[i] = f32_to_bf16_rne(v);
}

// out[n,:] = softmax(agg[n,:] + bias) over 32 channels
__global__ __launch_bounds__(256) void k_bias_softmax32(const float* __restrict__ agg,
                                                        const float* __restrict__ bias,
                                                        float* __restrict__ out, int Nn) {
  int n = blockIdx.x * blockDim.x + threadIdx.x;
  if (n >= Nn) return;
  float buf[32];
  float mx = -3.402823466e38f;
#pragma unroll
  for (int c = 0; c < 32; ++c) {
    float v = agg[(size_t)n * 32 + c] + bias[c];
    buf[c] = v;
    mx = v > mx ? v : mx;
  }
  float s = 0.f;
#pragma unroll
  for (int c = 0; c < 32; ++c) { float e = __expf(buf[c] - mx); buf[c] = e; s += e; }
  float inv = 1.f / s;
#pragma unroll
  for (int c = 0; c < 32; ++c) out[(size_t)n * 32 + c] = buf[c] * inv;
}

// ---------- host ----------
static inline char* ws_take(char*& p, size_t bytes) {
  char* r = p;
  p += (bytes + 255) & ~(size_t)255;
  return r;
}

extern "C" void kernel_launch(void* const* d_in, const int* in_sizes, int n_in,
                              void* d_out, int out_size, void* d_ws, size_t ws_size,
                              hipStream_t stream) {
  const float* x    = (const float*)d_in[0];
  const int*   ei   = (const int*)d_in[1];
  const float* W1   = (const float*)d_in[2];
  const float* aS1  = (const float*)d_in[3];
  const float* aD1  = (const float*)d_in[4];
  const float* b1   = (const float*)d_in[5];
  const float* W2   = (const float*)d_in[6];
  const float* aS2  = (const float*)d_in[7];
  const float* aD2  = (const float*)d_in[8];
  const float* b2   = (const float*)d_in[9];

  const int IN = 128, HID = 64, HEADS = 4, F1 = HEADS * HID /*256*/, OUT = 32;
  const int N  = in_sizes[0] / IN;
  const int E  = in_sizes[1] / 2;
  const int ET = E + N;

  // workspace layout
  char* p = (char*)d_ws;
  float*          xp     = (float*)ws_take(p, (size_t)N * F1 * 4);
  unsigned short* x_bf   = (unsigned short*)ws_take(p, (size_t)N * IN * 2);
  unsigned short* W1T    = (unsigned short*)ws_take(p, (size_t)F1 * IN * 2);
  float*          as1    = (float*)ws_take(p, (size_t)N * HEADS * 4);
  float*          ad1    = (float*)ws_take(p, (size_t)N * HEADS * 4);
  unsigned*       mkey1  = (unsigned*)ws_take(p, (size_t)N * HEADS * 4);
  float*          ssum1  = (float*)ws_take(p, (size_t)N * HEADS * 4);
  float*          mdec1  = (float*)ws_take(p, (size_t)N * HEADS * 4);
  float*          alpha1 = (float*)ws_take(p, (size_t)ET * HEADS * 4);
  float*          agg1   = (float*)ws_take(p, (size_t)N * F1 * 4);
  unsigned short* h_bf   = (unsigned short*)ws_take(p, (size_t)N * F1 * 2);
  unsigned short* W2T    = (unsigned short*)ws_take(p, (size_t)OUT * F1 * 2);
  float*          z2     = (float*)ws_take(p, (size_t)N * OUT * 4);
  float*          as2    = (float*)ws_take(p, (size_t)N * 4);
  float*          ad2    = (float*)ws_take(p, (size_t)N * 4);
  unsigned*       mkey2  = (unsigned*)ws_take(p, (size_t)N * 4);
  float*          ssum2  = (float*)ws_take(p, (size_t)N * 4);
  float*          mdec2  = (float*)ws_take(p, (size_t)N * 4);
  float*          alpha2 = (float*)ws_take(p, (size_t)ET * 4);
  float*          agg2   = (float*)ws_take(p, (size_t)N * OUT * 4);

  const int B = 256;

  // zero accumulators / max keys (encoded 0 == below -inf; self-loops guarantee updates)
  hipMemsetAsync(agg1, 0, (size_t)N * F1 * 4, stream);
  hipMemsetAsync(ssum1, 0, (size_t)N * HEADS * 4, stream);
  hipMemsetAsync(mkey1, 0, (size_t)N * HEADS * 4, stream);
  hipMemsetAsync(agg2, 0, (size_t)N * OUT * 4, stream);
  hipMemsetAsync(ssum2, 0, (size_t)N * 4, stream);
  hipMemsetAsync(mkey2, 0, (size_t)N * 4, stream);

  // ---- layer 1 ----
  k_f32_to_bf16<<<CDIV(N * IN, B), B, 0, stream>>>(x, x_bf, N * IN);
  k_transpose_bf16<<<CDIV(IN * F1, B), B, 0, stream>>>(W1, W1T, IN, F1);
  {
    int tiles = (N / 16) * (F1 / 16);
    k_gemm_wmma_bf16<<<CDIV(tiles, 4), 128, 0, stream>>>(x_bf, W1T, xp, N, F1, IN);
  }
  k_att_scores<<<CDIV(N * HEADS, B), B, 0, stream>>>(xp, aS1, aD1, as1, ad1, N, HEADS, HID);
  k_edge_logits_max<<<CDIV(ET, B), B, 0, stream>>>(ei, E, N, HEADS, as1, ad1, alpha1, mkey1);
  k_decode_max<<<CDIV(N * HEADS, B), B, 0, stream>>>(mkey1, mdec1, N * HEADS);
  k_edge_exp_sum<<<CDIV(ET, B), B, 0, stream>>>(ei, E, N, HEADS, mdec1, alpha1, ssum1);
  k_edge_norm<<<CDIV((long long)ET * HEADS, B), B, 0, stream>>>(ei, E, N, HEADS, ssum1, alpha1);
  {
    long long total = (long long)ET * (HEADS * HID / 4);
    k_edge_aggregate<<<(unsigned)CDIV(total, B), B, 0, stream>>>(ei, E, N, HEADS, HID, alpha1, xp, agg1);
  }
  k_bias_elu_bf16<<<CDIV(N * F1, B), B, 0, stream>>>(agg1, b1, h_bf, N, F1);

  // ---- layer 2 ----
  k_transpose_bf16<<<CDIV(F1 * OUT, B), B, 0, stream>>>(W2, W2T, F1, OUT);
  {
    int tiles = (N / 16) * (OUT / 16);
    k_gemm_wmma_bf16<<<CDIV(tiles, 4), 128, 0, stream>>>(h_bf, W2T, z2, N, OUT, F1);
  }
  k_att_scores<<<CDIV(N, B), B, 0, stream>>>(z2, aS2, aD2, as2, ad2, N, 1, OUT);
  k_edge_logits_max<<<CDIV(ET, B), B, 0, stream>>>(ei, E, N, 1, as2, ad2, alpha2, mkey2);
  k_decode_max<<<CDIV(N, B), B, 0, stream>>>(mkey2, mdec2, N);
  k_edge_exp_sum<<<CDIV(ET, B), B, 0, stream>>>(ei, E, N, 1, mdec2, alpha2, ssum2);
  k_edge_norm<<<CDIV((long long)ET, B), B, 0, stream>>>(ei, E, N, 1, ssum2, alpha2);
  {
    long long total = (long long)ET * (OUT / 4);
    k_edge_aggregate<<<(unsigned)CDIV(total, B), B, 0, stream>>>(ei, E, N, 1, OUT, alpha2, z2, agg2);
  }
  k_bias_softmax32<<<CDIV(N, B), B, 0, stream>>>(agg2, b2, (float*)d_out, N);
}
